// SpatioTemporalTransformerXL_1022202217191
// MI455X (gfx1250) — matrix operand order repaired and checked by reading, not compiled, not run on hardware
//
#include <hip/hip_runtime.h>
#include <hip/hip_bf16.h>

// ---------------------------------------------------------------------------
// SpatioTemporalTransformerXL forward for gfx1250 (MI455X).
// fp32 residual stream, bf16 WMMA GEMMs (v_wmma_f32_16x16x32_bf16),
// fused flash-style axial attention with online softmax.
// ---------------------------------------------------------------------------

typedef __bf16 bf16_t;
typedef bf16_t v16bf __attribute__((ext_vector_type(16)));
typedef float  v8f   __attribute__((ext_vector_type(8)));

#define NTOK   16384      // B*T*L = 2*32*256
#define DMODEL 768
#define FFWD   3072
#define NHEAD  12
#define HDIM   64
#define TDIM   32
#define LDIM   256
#define NLAYER 4

__device__ __forceinline__ unsigned short f2bf(float f) {
    unsigned int u = __float_as_uint(f);
    u += 0x7FFFu + ((u >> 16) & 1u);          // round-to-nearest-even
    return (unsigned short)(u >> 16);
}

// 16-bit A-matrix 16x32 fragment layout (ISA 7.12.2):
// VGPR v, lanes 0-15 hold K = {0..7 (v0-3), 16..23 (v4-7)}, lanes 16-31 +8.
// NOTE: for even k, (m,k) and (m,k+1) land in consecutive halfwords of the
// same lane -> staging can use 32-bit packed stores.
__device__ __forceinline__ int aFragIdx(int m /*0..15*/, int k /*0..31*/) {
    int lane = m + (((k & 15) >= 8) ? 16 : 0);
    int v    = ((k >= 16) ? 4 : 0) + ((k & 7) >> 1);
    int e    = v * 2 + (k & 1);
    return lane * 16 + e;
}
// 16-bit B-matrix 32x16 fragment layout (same K order as A): lanes 0-15 K={0..7,16..23},
// lanes 16-31 +8, N = lane%16 (so k/k+1 pairs are also contiguous).
__device__ __forceinline__ int bFragIdx(int k /*0..31*/, int n /*0..15*/) {
    int lane = n + (((k & 15) >= 8) ? 16 : 0);
    return lane * 16 + ((k >= 16) ? 8 : 0) + (k & 7);
}

// gelu(x) = x * sigmoid(2*c0*(x + 0.044715 x^3)); single v_exp_f32, no branches.
__device__ __forceinline__ float geluf(float x) {
    const float c0 = 0.7978845608028654f;     // sqrt(2/pi)
    float t = c0 * (x + 0.044715f * x * x * x);
    return x / (1.f + __expf(-2.f * t));
}

// ---------------------------------------------------------------------------
// Block reduction (wave32 shuffle + 8-partial LDS combine)
// ---------------------------------------------------------------------------
__device__ __forceinline__ float blockReduceSum(float s, float* red) {
    const int lane = threadIdx.x & 31, wave = threadIdx.x >> 5;
#pragma unroll
    for (int off = 16; off > 0; off >>= 1) s += __shfl_xor(s, off, 32);
    __syncthreads();                           // protect red across calls
    if (lane == 0) red[wave] = s;
    __syncthreads();
    float t = 0.f;
#pragma unroll
    for (int i = 0; i < 8; ++i) t += red[i];
    return t;
}

// ---------------------------------------------------------------------------
// LayerNorm (fp32 in) -> bf16 out (scale/offset folded in)
// ---------------------------------------------------------------------------
__global__ __launch_bounds__(256) void layernorm_bf16_kernel(
    const float* __restrict__ h, const float* __restrict__ scale,
    const float* __restrict__ bias, unsigned short* __restrict__ out)
{
    __shared__ float red[8];
    const int row = blockIdx.x, tid = threadIdx.x;
    const float* x = h + (size_t)row * DMODEL;

    float x0 = x[tid], x1 = x[tid + 256], x2 = x[tid + 512];
    const float mean = blockReduceSum(x0 + x1 + x2, red) * (1.f / DMODEL);
    float d0 = x0 - mean, d1 = x1 - mean, d2 = x2 - mean;
    const float var = blockReduceSum(d0 * d0 + d1 * d1 + d2 * d2, red) * (1.f / DMODEL);
    const float inv = rsqrtf(var + 1e-5f);

    size_t o = (size_t)row * DMODEL;
    out[o + tid]       = f2bf(d0 * inv * scale[tid]       + bias[tid]);
    out[o + tid + 256] = f2bf(d1 * inv * scale[tid + 256] + bias[tid + 256]);
    out[o + tid + 512] = f2bf(d2 * inv * scale[tid + 512] + bias[tid + 512]);
}

// ---------------------------------------------------------------------------
// GEMM: C[N,Nout] = A_bf16[N,K] * W_f32->bf16[K,Nout] (+bias) (+epilogue)
// 128x64 block tile, BK=32. 8 waves: wave w owns one 16-row A fragment and
// all 4 N-subtiles -> 4 WMMAs per k-step per wave.
// epi: 0 = store bf16, 1 = gelu->bf16, 2 = accumulate into fp32 residual.
// ---------------------------------------------------------------------------
__global__ __launch_bounds__(256) void gemm_bf16_kernel(
    const unsigned short* __restrict__ A, const float* __restrict__ W,
    const float* __restrict__ bias, unsigned short* __restrict__ outB,
    float* __restrict__ outR, int N, int K, int Nout, int epi)
{
    __shared__ __attribute__((aligned(32))) unsigned short sA[8 * 512]; // 128x32
    __shared__ __attribute__((aligned(32))) unsigned short sB[4 * 512]; // 32x64

    const int tid = threadIdx.x, lane = tid & 31, wave = tid >> 5;
    const int n0 = blockIdx.x * 64, m0 = blockIdx.y * 128;

    v8f acc[4] = {};
    for (int k0 = 0; k0 < K; k0 += 32) {
        // Prefetch next k-tile into cache while this one is staged/consumed.
        if (k0 + 32 < K) {
            __builtin_prefetch(A + (size_t)(m0 + (tid & 127)) * K + (k0 + 32), 0, 3);
            if (tid < 32)
                __builtin_prefetch(W + (size_t)(k0 + 32 + tid) * Nout + n0, 0, 3);
        }
        // Stage A: 128x32 bf16, packed 2 halves per 32-bit load/store.
#pragma unroll
        for (int i = 0; i < 8; ++i) {
            int p = i * 256 + tid;
            int m = p >> 4, kp = (p & 15) * 2;
            unsigned int val = *(const unsigned int*)(A + (size_t)(m0 + m) * K + (k0 + kp));
            *(unsigned int*)(sA + (m >> 4) * 512 + aFragIdx(m & 15, kp)) = val;
        }
        // Stage B: 32x64 fp32 -> bf16, packed pairs along k.
#pragma unroll
        for (int i = 0; i < 4; ++i) {
            int p = i * 256 + tid;
            int n = p & 63, kp = (p >> 6) * 2;
            float w0 = W[(size_t)(k0 + kp) * Nout + (n0 + n)];
            float w1 = W[(size_t)(k0 + kp + 1) * Nout + (n0 + n)];
            unsigned int val = (unsigned int)f2bf(w0) | ((unsigned int)f2bf(w1) << 16);
            *(unsigned int*)(sB + (n >> 4) * 512 + bFragIdx(kp, n & 15)) = val;
        }
        __syncthreads();
        v16bf af = *(const v16bf*)(sA + wave * 512 + lane * 16);
#pragma unroll
        for (int j = 0; j < 4; ++j) {
            v16bf bf = *(const v16bf*)(sB + j * 512 + lane * 16);
            acc[j] = __builtin_amdgcn_wmma_f32_16x16x32_bf16(false, af, false, bf,
                                                             (short)0, acc[j], false, false);
        }
        __syncthreads();
    }

    const int hi8 = (lane >= 16) ? 8 : 0;
#pragma unroll
    for (int j = 0; j < 4; ++j) {
        int n = n0 + j * 16 + (lane & 15);
        float bb = bias ? bias[n] : 0.f;
#pragma unroll
        for (int i = 0; i < 8; ++i) {
            int m = m0 + wave * 16 + i + hi8;                  // C/D layout
            float val = acc[j][i] + bb;
            if (epi == 1) val = geluf(val);
            size_t o = (size_t)m * Nout + n;
            if (epi == 2) outR[o] += val;
            else          outB[o] = f2bf(val);
        }
    }
}

// ---------------------------------------------------------------------------
// Fused axial attention (flash-style, online softmax).
// One 128-thread block = one (instance, head, 32-query tile).
// mode 0 = time (seq over T, row stride L); mode 1 = space (seq over L).
// ---------------------------------------------------------------------------
__global__ __launch_bounds__(128) void attn_kernel(
    const unsigned short* __restrict__ q, const unsigned short* __restrict__ k,
    const unsigned short* __restrict__ v, unsigned short* __restrict__ out, int mode)
{
    __shared__ __attribute__((aligned(32))) unsigned short sQ[4 * 512]; // [msub][kstep]
    __shared__ __attribute__((aligned(32))) unsigned short sK[4 * 512]; // [kstep][nsub]
    __shared__ __attribute__((aligned(32))) unsigned short sV[4 * 512]; // [nsub] (K=32)
    __shared__ __attribute__((aligned(32))) unsigned short sP[2 * 512]; // [msub]
    __shared__ float sS[32 * 32];
    __shared__ float sM[32], sLsum[32], sAlpha[32];

    const int inst = blockIdx.z, head = blockIdx.y, qt = blockIdx.x;
    int base, strideRow, seqLen;
    if (mode == 0) { int b = inst / LDIM, ll = inst % LDIM;
                     base = b * TDIM * LDIM + ll; strideRow = LDIM; seqLen = TDIM; }
    else           { base = inst * LDIM; strideRow = 1; seqLen = LDIM; }

    const int tid = threadIdx.x, lane = tid & 31, wave = tid >> 5;

    // Stage Q tile (32x64) into A-fragment order, once (packed pairs along d).
#pragma unroll
    for (int i = 0; i < 8; ++i) {
        int p = i * 128 + tid;
        int mq = p >> 5, dp = (p & 31) * 2;
        int g = base + (qt * 32 + mq) * strideRow;
        unsigned int val = *(const unsigned int*)(q + (size_t)g * DMODEL + head * HDIM + dp);
        *(unsigned int*)(sQ + ((mq >> 4) * 2 + (dp >> 5)) * 512 +
                         aFragIdx(mq & 15, dp & 31)) = val;
    }

    v8f oacc0 = {}, oacc1 = {};
    const int numChunks = seqLen / 32;
    for (int c = 0; c < numChunks; ++c) {
        // Stage K chunk (B-frags for S=Q*K^T) and V chunk (B-frags for O+=P*V).
#pragma unroll
        for (int i = 0; i < 16; ++i) {
            int idx = i * 128 + tid;
            int j = idx >> 6, d = idx & 63;
            int g = base + (c * 32 + j) * strideRow;
            size_t o = (size_t)g * DMODEL + head * HDIM + d;
            sK[((d >> 5) * 2 + (j >> 4)) * 512 + bFragIdx(d & 31, j & 15)] = k[o];
            sV[(d >> 4) * 512 + bFragIdx(j, d & 15)] = v[o];
        }
        __syncthreads();

        { // S = Q*K^T * 1/sqrt(64): 4 waves x one 16x16 subtile, 2 K-steps.
            int mi = wave >> 1, ni = wave & 1;
            v8f sacc = {};
#pragma unroll
            for (int ks = 0; ks < 2; ++ks) {
                v16bf aq = *(const v16bf*)(sQ + (mi * 2 + ks) * 512 + lane * 16);
                v16bf bk = *(const v16bf*)(sK + (ks * 2 + ni) * 512 + lane * 16);
                sacc = __builtin_amdgcn_wmma_f32_16x16x32_bf16(false, aq, false, bk,
                                                               (short)0, sacc, false, false);
            }
            int hi8 = (lane >= 16) ? 8 : 0;
#pragma unroll
            for (int i = 0; i < 8; ++i)
                sS[(mi * 16 + i + hi8) * 32 + ni * 16 + (lane & 15)] = sacc[i] * 0.125f;
        }
        __syncthreads();

        // Online softmax per query row (threads 0..31).
        if (tid < 32) {
            int r = tid;
            float mOld = (c == 0) ? -1e30f : sM[r];
            float lOld = (c == 0) ? 0.f : sLsum[r];
            float mNew = mOld;
            for (int j = 0; j < 32; ++j) mNew = fmaxf(mNew, sS[r * 32 + j]);
            float alpha = __expf(mOld - mNew);
            float lsum = 0.f;
            for (int j = 0; j < 32; ++j) {
                float p = __expf(sS[r * 32 + j] - mNew);
                lsum += p;
                sP[(r >> 4) * 512 + aFragIdx(r & 15, j)] = f2bf(p);
            }
            sM[r] = mNew; sLsum[r] = lOld * alpha + lsum; sAlpha[r] = alpha;
        }
        __syncthreads();

        { // Rescale running O and accumulate P*V: each wave owns 2 subtiles.
            int mo = wave >> 1, no = (wave & 1) * 2;
            int hi8 = (lane >= 16) ? 8 : 0;
#pragma unroll
            for (int i = 0; i < 8; ++i) {
                float a = sAlpha[mo * 16 + i + hi8];
                oacc0[i] *= a; oacc1[i] *= a;
            }
            v16bf pf  = *(const v16bf*)(sP + mo * 512 + lane * 16);
            v16bf vf0 = *(const v16bf*)(sV + (no + 0) * 512 + lane * 16);
            v16bf vf1 = *(const v16bf*)(sV + (no + 1) * 512 + lane * 16);
            oacc0 = __builtin_amdgcn_wmma_f32_16x16x32_bf16(false, pf, false, vf0,
                                                            (short)0, oacc0, false, false);
            oacc1 = __builtin_amdgcn_wmma_f32_16x16x32_bf16(false, pf, false, vf1,
                                                            (short)0, oacc1, false, false);
        }
        __syncthreads();
    }

    // Finalize: O /= l, store bf16 into attention-output buffer.
    {
        int mo = wave >> 1, no = (wave & 1) * 2;
        int hi8 = (lane >= 16) ? 8 : 0;
        int col0 = head * HDIM + no * 16 + (lane & 15);
#pragma unroll
        for (int i = 0; i < 8; ++i) {
            int r = mo * 16 + i + hi8;
            float inv = 1.f / sLsum[r];
            int g = base + (qt * 32 + r) * strideRow;
            size_t o = (size_t)g * DMODEL;
            out[o + col0]      = f2bf(oacc0[i] * inv);
            out[o + col0 + 16] = f2bf(oacc1[i] * inv);
        }
    }
}

// ---------------------------------------------------------------------------
// Host orchestration
// ---------------------------------------------------------------------------
extern "C" void kernel_launch(void* const* d_in, const int* in_sizes, int n_in,
                              void* d_out, int out_size, void* d_ws, size_t ws_size,
                              hipStream_t stream) {
    (void)in_sizes; (void)n_in; (void)out_size; (void)ws_size;
    const float* emb  = (const float*)d_in[0];
    const float* Wq_t = (const float*)d_in[1];
    const float* Wk_t = (const float*)d_in[2];
    const float* Wv_t = (const float*)d_in[3];
    const float* Wo_t = (const float*)d_in[4];
    const float* bo_t = (const float*)d_in[5];
    const float* Wq_s = (const float*)d_in[6];
    const float* Wk_s = (const float*)d_in[7];
    const float* Wv_s = (const float*)d_in[8];
    const float* Wo_s = (const float*)d_in[9];
    const float* bo_s = (const float*)d_in[10];
    const float* lnts = (const float*)d_in[11];
    const float* lntb = (const float*)d_in[12];
    const float* lnss = (const float*)d_in[13];
    const float* lnsb = (const float*)d_in[14];
    const float* ln2s = (const float*)d_in[15];
    const float* ln2b = (const float*)d_in[16];
    const float* W1   = (const float*)d_in[17];
    const float* b1   = (const float*)d_in[18];
    const float* W2   = (const float*)d_in[19];
    const float* b2   = (const float*)d_in[20];

    const int N = NTOK, D = DMODEL, F = FFWD;
    float* h = (float*)d_out;                                // fp32 residual stream
    unsigned short* xln = (unsigned short*)d_ws;             // N*D bf16 (LN out / attn out)
    unsigned short* big = xln + (size_t)N * D;               // N*F bf16 region
    unsigned short* qb  = big;                               //   q  (N*D)
    unsigned short* kb  = big + (size_t)N * D;               //   k  (N*D)
    unsigned short* vb  = big + 2 * (size_t)N * D;           //   v  (N*D)
    unsigned short* ffh = big;                               //   reused as FFN hidden (N*F)

    hipMemcpyAsync(h, emb, (size_t)N * D * sizeof(float), hipMemcpyDeviceToDevice, stream);

    const dim3 gD(D / 64, N / 128), gF(F / 64, N / 128), b256(256);
    for (int l = 0; l < NLAYER; ++l) {
        const size_t wDD = (size_t)l * D * D, wDF = (size_t)l * D * F;
        const size_t vD = (size_t)l * D, vF = (size_t)l * F;
        // ---- time attention + residual ----
        layernorm_bf16_kernel<<<N, b256, 0, stream>>>(h, lnts + vD, lntb + vD, xln);
        gemm_bf16_kernel<<<gD, b256, 0, stream>>>(xln, Wq_t + wDD, nullptr, qb, nullptr, N, D, D, 0);
        gemm_bf16_kernel<<<gD, b256, 0, stream>>>(xln, Wk_t + wDD, nullptr, kb, nullptr, N, D, D, 0);
        gemm_bf16_kernel<<<gD, b256, 0, stream>>>(xln, Wv_t + wDD, nullptr, vb, nullptr, N, D, D, 0);
        attn_kernel<<<dim3(1, NHEAD, 2 * LDIM), 128, 0, stream>>>(qb, kb, vb, xln, 0);
        gemm_bf16_kernel<<<gD, b256, 0, stream>>>(xln, Wo_t + wDD, bo_t + vD, nullptr, h, N, D, D, 2);
        // ---- space attention + residual ----
        layernorm_bf16_kernel<<<N, b256, 0, stream>>>(h, lnss + vD, lnsb + vD, xln);
        gemm_bf16_kernel<<<gD, b256, 0, stream>>>(xln, Wq_s + wDD, nullptr, qb, nullptr, N, D, D, 0);
        gemm_bf16_kernel<<<gD, b256, 0, stream>>>(xln, Wk_s + wDD, nullptr, kb, nullptr, N, D, D, 0);
        gemm_bf16_kernel<<<gD, b256, 0, stream>>>(xln, Wv_s + wDD, nullptr, vb, nullptr, N, D, D, 0);
        attn_kernel<<<dim3(LDIM / 32, NHEAD, 2 * TDIM), 128, 0, stream>>>(qb, kb, vb, xln, 1);
        gemm_bf16_kernel<<<gD, b256, 0, stream>>>(xln, Wo_s + wDD, bo_s + vD, nullptr, h, N, D, D, 2);
        // ---- FFN + residual ----
        layernorm_bf16_kernel<<<N, b256, 0, stream>>>(h, ln2s + vD, ln2b + vD, xln);
        gemm_bf16_kernel<<<gF, b256, 0, stream>>>(xln, W1 + wDF, b1 + vF, ffh, nullptr, N, D, F, 1);
        gemm_bf16_kernel<<<gD, b256, 0, stream>>>(ffh, W2 + wDF, b2 + vD, nullptr, h, N, F, D, 2);
    }
}
